// BiLSTM_13142599926170
// MI455X (gfx1250) — matrix operand order repaired and checked by reading, not compiled, not run on hardware
//
#include <hip/hip_runtime.h>
#include <hip/hip_bf16.h>

typedef __attribute__((ext_vector_type(16))) __bf16 bf16x16;
typedef __attribute__((ext_vector_type(8)))  __bf16 bf16x8;
typedef __attribute__((ext_vector_type(8)))  float  v8f;

#define B_  64
#define T_  512
#define D_  256
#define H_  256
#define NG  1024            // 4*H gate width
#define BT  (B_ * T_)

// ---------------------------------------------------------------------------
// A-fragment loader: 16x32 bf16 A tile from a row-major matrix.
// Per ISA layout: lane(0-15)=row m, half=lane>>4; elems 0-7 <- k0+half*8..+7,
// elems 8-15 <- k0+16+half*8..+7  (two contiguous 16-byte loads).
// ---------------------------------------------------------------------------
__device__ __forceinline__ bf16x16 load_a_frag(const __bf16* rowk, int half) {
    bf16x8 lo = *(const bf16x8*)(rowk + half * 8);
    bf16x8 hi = *(const bf16x8*)(rowk + 16 + half * 8);
    bf16x16 a;
#pragma unroll
    for (int i = 0; i < 8; ++i) { a[i] = lo[i]; a[i + 8] = hi[i]; }
    return a;
}

__device__ __forceinline__ v8f wmma_bf16(bf16x16 a, bf16x16 b, v8f c) {
    return __builtin_amdgcn_wmma_f32_16x16x32_bf16(false, a, false, b, (short)0, c,
                                                   false, false);
}

__device__ __forceinline__ float sigmoidf_(float x) {
    return 1.0f / (1.0f + __expf(-x));
}

// ---------------------------------------------------------------------------
// Pack W [N=1024][K] row-major f32 -> bf16 WMMA B-tiles.
// B tile (32K x 16N): lane(0-15)=col n, half=lane>>4, elem e -> k = half*8+(e&7)+(e>>3)*16.
// Packed layout: [ntile][ktile][lane*16+e], so dst[idx] with idx linear.
// ---------------------------------------------------------------------------
__global__ void pack_b_kernel(const float* __restrict__ W, __bf16* __restrict__ dst,
                              int K) {
    int idx  = blockIdx.x * 256 + threadIdx.x;        // over N*K
    int e    = idx & 15;
    int lane = (idx >> 4) & 31;
    int tile = idx >> 9;
    int KT   = K >> 5;
    int kt   = tile % KT;
    int nt   = tile / KT;
    int n    = nt * 16 + (lane & 15);
    int k    = kt * 32 + (lane >> 4) * 8 + (e & 7) + (e >> 3) * 16;
    dst[idx] = (__bf16)W[(size_t)n * K + k];
}

__global__ void bias_kernel(const float* __restrict__ a, const float* __restrict__ b,
                            float* __restrict__ d) {
    int i = blockIdx.x * 256 + threadIdx.x;
    if (i < NG) d[i] = a[i] + b[i];
}

__global__ void rev_kernel(const int* __restrict__ sentlen, int* __restrict__ rev) {
    int idx = blockIdx.x * 256 + threadIdx.x;          // over B*T
    int b = idx >> 9, t = idx & 511;
    int len = sentlen[b];
    rev[idx] = (t < len) ? (len - 1 - t) : t;
}

__global__ void embed_kernel(const int* __restrict__ x, const float* __restrict__ embed,
                             __bf16* __restrict__ emb) {
    int idx = blockIdx.x * 256 + threadIdx.x;          // over B*T*D/4
    int row = idx >> 6;                                // D/4 = 64 per row
    int c   = (idx & 63) * 4;
    int tok = x[row];
    float4 v = *(const float4*)(embed + (size_t)tok * D_ + c);
    __bf16* o = emb + (size_t)row * D_ + c;
    o[0] = (__bf16)v.x; o[1] = (__bf16)v.y; o[2] = (__bf16)v.z; o[3] = (__bf16)v.w;
}

// ---------------------------------------------------------------------------
// Input-projection GEMM: Xp[M][1024] = A[M][K](bf16) x W^T + bias (f32 out).
// One wave computes a 16x64 strip (A frag reused over 4 N-tiles).
// ---------------------------------------------------------------------------
__global__ __launch_bounds__(256) void gemm_xp_kernel(
    const __bf16* __restrict__ A, const __bf16* __restrict__ Bpk,
    const float* __restrict__ bias, float* __restrict__ Xp, int K) {
    const int KT    = K >> 5;
    const int wavei = threadIdx.x >> 5;
    const int lane  = threadIdx.x & 31;
    const int strip = blockIdx.x * 8 + wavei;          // (M/16)*16 strips
    const int mtile = strip >> 4;
    const int nbase = (strip & 15) * 4;                // 4 consecutive n-tiles
    const int lrow  = lane & 15, half = lane >> 4;
    const int row   = mtile * 16 + lrow;

    v8f acc[4];
#pragma unroll
    for (int j = 0; j < 4; ++j)
#pragma unroll
        for (int r = 0; r < 8; ++r) acc[j][r] = 0.0f;

    for (int kt = 0; kt < KT; ++kt) {
        bf16x16 a = load_a_frag(A + (size_t)row * K + kt * 32, half);
#pragma unroll
        for (int j = 0; j < 4; ++j) {
            bf16x16 b = *(const bf16x16*)(Bpk + ((size_t)(nbase + j) * KT + kt) * 512 +
                                          lane * 16);
            acc[j] = wmma_bf16(a, b, acc[j]);
        }
    }
#pragma unroll
    for (int j = 0; j < 4; ++j)
#pragma unroll
        for (int r = 0; r < 8; ++r) {
            int m = mtile * 16 + r + half * 8;
            int n = (nbase + j) * 16 + lrow;
            Xp[(size_t)m * NG + n] = acc[j][r] + bias[n];
        }
}

// ---------------------------------------------------------------------------
// Persistent LSTM recurrence. grid=2 (dir), 1024 threads = 32 waves.
// h (bf16) lives in LDS; cell state c lives in VGPRs (each wave owns a
// disjoint 32-batch x 16-j region and computes the matching i/f/g/o tiles,
// so no cross-wave gate exchange is needed). 2 barriers per timestep.
// Backward dir reads xp rows / writes outputs through rev_idx (length-aware
// reversal == the reference's pack_padded semantics); the 64-entry rev row
// for step t+1 is staged into LDS during step t, and next-step xp gate rows
// are prefetched (global_prefetch_b8) so the serial gather latency hides
// under the WMMA k-loop.
// ---------------------------------------------------------------------------
__global__ __launch_bounds__(1024) void lstm_rec_kernel(
    const float* __restrict__ xpF, const float* __restrict__ xpB,
    const __bf16* __restrict__ whhF, const __bf16* __restrict__ whhB,
    const int* __restrict__ rev,
    __bf16* __restrict__ outb, float* __restrict__ outf) {
    const int dir = blockIdx.x;
    const float*  xp  = dir ? xpB : xpF;
    const __bf16* whh = dir ? whhB : whhF;
    const int colOff  = dir * H_;

    __shared__ __align__(16) __bf16 hbuf[B_][H_];      // 32 KB
    __shared__ int revrow[2][B_];                      // staged rev_idx rows

    const int tid   = threadIdx.x;
    const int wavei = tid >> 5, lane = tid & 31;
    const int mhalf = wavei >> 4;                      // batch rows 0..31 / 32..63
    const int jt    = wavei & 15;                      // 16-wide j tile
    const int lrow  = lane & 15, half = lane >> 4;

    for (int i = tid; i < B_ * H_; i += 1024) (&hbuf[0][0])[i] = (__bf16)0.0f;
    if (dir && tid < B_) revrow[0][tid] = rev[tid * T_];   // row for t=0

    v8f creg[2];
#pragma unroll
    for (int mt = 0; mt < 2; ++mt)
#pragma unroll
        for (int r = 0; r < 8; ++r) creg[mt][r] = 0.0f;

    __syncthreads();

    for (int t = 0; t < T_; ++t) {
        // stage rev row for step t+1 (latency hidden under the k-loop below)
        if (dir && tid < B_) {
            int tn = (t + 1 < T_) ? (t + 1) : t;
            revrow[(t + 1) & 1][tid] = rev[tid * T_ + tn];
        }

        v8f acc[4][2];
#pragma unroll
        for (int g = 0; g < 4; ++g)
#pragma unroll
            for (int mt = 0; mt < 2; ++mt)
#pragma unroll
                for (int r = 0; r < 8; ++r) acc[g][mt][r] = 0.0f;

#pragma unroll
        for (int kt = 0; kt < H_ / 32; ++kt) {
            bf16x16 a0 = load_a_frag(&hbuf[mhalf * 32 + lrow][kt * 32], half);
            bf16x16 a1 = load_a_frag(&hbuf[mhalf * 32 + 16 + lrow][kt * 32], half);
#pragma unroll
            for (int g = 0; g < 4; ++g) {
                int nt = g * 16 + jt;
                bf16x16 b = *(const bf16x16*)(whh + ((size_t)nt * 8 + kt) * 512 +
                                              lane * 16);
                acc[g][0] = wmma_bf16(a0, b, acc[g][0]);
                acc[g][1] = wmma_bf16(a1, b, acc[g][1]);
            }
        }
        __syncthreads();     // hbuf reads done; staged revrow[(t+1)&1] visible

        // prefetch next step's xp gate segments (one batch-row per lane)
        if (t + 1 < T_) {
            int bRowp = mhalf * 32 + lane;
            int posn  = dir ? revrow[(t + 1) & 1][bRowp] : (t + 1);
            const float* pf = xp + ((size_t)bRowp * T_ + posn) * NG + jt * 16;
#pragma unroll
            for (int g = 0; g < 4; ++g)
                __builtin_prefetch(pf + g * H_, 0, 3);
        }

#pragma unroll
        for (int mt = 0; mt < 2; ++mt)
#pragma unroll
            for (int r = 0; r < 8; ++r) {
                int bRow = mhalf * 32 + mt * 16 + r + half * 8;
                int j    = jt * 16 + lrow;
                int pos  = dir ? revrow[t & 1][bRow] : t;
                size_t rowbase = ((size_t)bRow * T_ + pos) * NG;
                float gi = acc[0][mt][r] + xp[rowbase + 0 * H_ + j];
                float gf = acc[1][mt][r] + xp[rowbase + 1 * H_ + j];
                float gg = acc[2][mt][r] + xp[rowbase + 2 * H_ + j];
                float go = acc[3][mt][r] + xp[rowbase + 3 * H_ + j];
                float cv = sigmoidf_(gf) * creg[mt][r] + sigmoidf_(gi) * tanhf(gg);
                creg[mt][r] = cv;
                float h = sigmoidf_(go) * tanhf(cv);
                hbuf[bRow][j] = (__bf16)h;
                size_t orow = (size_t)bRow * T_ + pos;
                if (outb) outb[orow * (2 * H_) + colOff + j] = (__bf16)h;
                else      outf[orow * (2 * H_) + colOff + j] = h;
            }
        __syncthreads();     // h writes done before next step's reads
    }
}

__global__ void maxpool_kernel(const float* __restrict__ h2,
                               const int* __restrict__ sentlen,
                               float* __restrict__ out) {
    int idx = blockIdx.x * 256 + threadIdx.x;          // over B*2H = 32768
    int b = idx >> 9, n = idx & 511;
    int len = sentlen[b];
    float m = -__builtin_inff();
    const float* p = h2 + (size_t)b * T_ * (2 * H_) + n;
    for (int t = 0; t < len; ++t) m = fmaxf(m, p[(size_t)t * (2 * H_)]);
    out[idx] = m;
}

// ---------------------------------------------------------------------------

extern "C" void kernel_launch(void* const* d_in, const int* in_sizes, int n_in,
                              void* d_out, int out_size, void* d_ws, size_t ws_size,
                              hipStream_t stream) {
    const int*   x       = (const int*)d_in[0];
    const int*   sentlen = (const int*)d_in[1];
    const float* embed   = (const float*)d_in[2];
    const float* Wih[4], *Whh[4], *bih[4], *bhh[4];    // order: 0f,0b,1f,1b
    for (int i = 0; i < 4; ++i) {
        Wih[i] = (const float*)d_in[3 + i * 4 + 0];
        Whh[i] = (const float*)d_in[3 + i * 4 + 1];
        bih[i] = (const float*)d_in[3 + i * 4 + 2];
        bhh[i] = (const float*)d_in[3 + i * 4 + 3];
    }

    char* ws = (char*)d_ws;
    size_t off = 0;
    auto alloc = [&](size_t n) { size_t o = off; off += (n + 255) & ~(size_t)255; return o; };

    __bf16* emb   = (__bf16*)(ws + alloc((size_t)BT * D_ * 2));        // 16 MB
    __bf16* h1    = (__bf16*)(ws + alloc((size_t)BT * 2 * H_ * 2));    // 32 MB
    float*  h2    = (float*) (ws + alloc((size_t)BT * 2 * H_ * 4));    // 64 MB
    float*  xpF   = (float*) (ws + alloc((size_t)BT * NG * 4));        // 128 MB
    float*  xpB   = (float*) (ws + alloc((size_t)BT * NG * 4));        // 128 MB
    __bf16* pWih[4]; __bf16* pWhh[4]; float* pBias[4];
    const int Kin[4] = {D_, D_, 2 * H_, 2 * H_};
    for (int i = 0; i < 4; ++i) {
        pWih[i]  = (__bf16*)(ws + alloc((size_t)NG * Kin[i] * 2));
        pWhh[i]  = (__bf16*)(ws + alloc((size_t)NG * H_ * 2));
        pBias[i] = (float*) (ws + alloc((size_t)NG * 4));
    }
    int* rev = (int*)(ws + alloc((size_t)BT * 4));

    // --- prep: pack weights, fold biases, rev_idx, embedding gather (bf16) ---
    for (int i = 0; i < 4; ++i) {
        pack_b_kernel<<<(NG * Kin[i]) / 256, 256, 0, stream>>>(Wih[i], pWih[i], Kin[i]);
        pack_b_kernel<<<(NG * H_) / 256, 256, 0, stream>>>(Whh[i], pWhh[i], H_);
        bias_kernel<<<4, 256, 0, stream>>>(bih[i], bhh[i], pBias[i]);
    }
    rev_kernel<<<BT / 256, 256, 0, stream>>>(sentlen, rev);
    embed_kernel<<<(BT * D_ / 4) / 256, 256, 0, stream>>>(x, embed, emb);

    const int gemmBlocks = (BT / 16) * 16 / 8;         // 4096

    // --- layer 0 ---
    gemm_xp_kernel<<<gemmBlocks, 256, 0, stream>>>(emb, pWih[0], pBias[0], xpF, D_);
    gemm_xp_kernel<<<gemmBlocks, 256, 0, stream>>>(emb, pWih[1], pBias[1], xpB, D_);
    lstm_rec_kernel<<<2, 1024, 0, stream>>>(xpF, xpB, pWhh[0], pWhh[1], rev,
                                            h1, (float*)nullptr);

    // --- layer 1 (reuse xp buffers) ---
    gemm_xp_kernel<<<gemmBlocks, 256, 0, stream>>>(h1, pWih[2], pBias[2], xpF, 2 * H_);
    gemm_xp_kernel<<<gemmBlocks, 256, 0, stream>>>(h1, pWih[3], pBias[3], xpB, 2 * H_);
    lstm_rec_kernel<<<2, 1024, 0, stream>>>(xpF, xpB, pWhh[2], pWhh[3], rev,
                                            (__bf16*)nullptr, h2);

    // --- ragged max-pool ---
    maxpool_kernel<<<(B_ * 2 * H_) / 256, 256, 0, stream>>>(h2, sentlen, (float*)d_out);
}
